// SimpleLSTM_63136019251791
// MI455X (gfx1250) — compile-verified, matrix operand order checked by reading, not run on hardware
//
#include <hip/hip_runtime.h>

// ---------------------------------------------------------------------------
// 2-layer LSTM, persistent-kernel implementation for gfx1250 (MI455X).
// B=64, T=512, I=256, H=512.  bf16 WMMA (v_wmma_f32_16x16x32_bf16), fp32 accum.
// ---------------------------------------------------------------------------

typedef float  v8f   __attribute__((ext_vector_type(8)));
typedef __bf16 v16bf __attribute__((ext_vector_type(16)));
typedef __bf16 v8bf  __attribute__((ext_vector_type(8)));

constexpr int Bb = 64, Tt = 512, Ii = 256, Hh = 512;
constexpr int NWG = Hh / 16;        // 32 workgroups, one 16-column H slice each
constexpr int NWAVE = Bb / 16;      // 4 waves, one 16-row batch tile each
constexpr int THREADS = NWAVE * 32; // 128 threads (wave32)

// LDS staging sizes (elements of bf16) — per-WG weight slices, 4 gate chunks x 16 rows
constexpr int WX0_E = 4 * 16 * Ii;  // 16384
constexpr int WH0_E = 4 * 16 * Hh;  // 32768
constexpr int WX1_E = 4 * 16 * Hh;  // 32768
constexpr int WH1_E = 4 * 16 * Hh;  // 32768
constexpr int LDS_E = WX0_E + WH0_E + WX1_E + WH1_E; // 114688 elems = 224 KB

__device__ __forceinline__ unsigned short f2bf(float f) {
    unsigned u = __float_as_uint(f);
    unsigned r = (u + 0x7FFFu + ((u >> 16) & 1u)) >> 16;   // round-to-nearest-even
    return (unsigned short)r;
}

__device__ __forceinline__ float sigf(float v) {
    float e = __expf(-v);
    return __builtin_amdgcn_rcpf(1.0f + e);
}
__device__ __forceinline__ float tanh_fast(float v) {
    float e = __expf(-2.0f * v);
    return (1.0f - e) * __builtin_amdgcn_rcpf(1.0f + e);
}

// Build a 16x32 WMMA A/B fragment for this lane from a row pointer at k-offset 0.
// ISA 7.12.2 (16-bit A 16x32): lane group g = lane/16 holds K = 8g..8g+7 (VGPR0-3)
// and K = 16+8g..16+8g+7 (VGPR4-7).  Two contiguous 16-byte loads.
__device__ __forceinline__ v16bf frag(const unsigned short* p, int g) {
    v8bf lo = *(const v8bf*)(const void*)(p + 8 * g);
    v8bf hi = *(const v8bf*)(const void*)(p + 16 + 8 * g);
    v16bf r;
#pragma unroll
    for (int i = 0; i < 8; ++i) { r[i] = lo[i]; r[i + 8] = hi[i]; }
    return r;
}

__device__ __forceinline__ v8f wmma_bf16(v16bf a, v16bf b, v8f c) {
    return __builtin_amdgcn_wmma_f32_16x16x32_bf16(false, a, false, b, (short)0, c,
                                                   false, false);
}

// Device-wide barrier: 32 resident WGs, agent-scope atomics, s_sleep spin.
__device__ __forceinline__ void grid_sync(unsigned* cnt, unsigned target) {
    __builtin_amdgcn_fence(__ATOMIC_RELEASE, "agent");
    __syncthreads();
    if (threadIdx.x == 0) {
        __hip_atomic_fetch_add(cnt, 1u, __ATOMIC_RELAXED, __HIP_MEMORY_SCOPE_AGENT);
        while (__hip_atomic_load(cnt, __ATOMIC_RELAXED, __HIP_MEMORY_SCOPE_AGENT) < target)
            __builtin_amdgcn_s_sleep(2);
    }
    __syncthreads();
    __builtin_amdgcn_fence(__ATOMIC_ACQUIRE, "agent");
}

// ------------------------------ prep kernels -------------------------------

__global__ void prep_x_kernel(const float* __restrict__ x,
                              unsigned short* __restrict__ xbf, int n) {
    int i = blockIdx.x * blockDim.x + threadIdx.x;
    if (i < n) xbf[i] = f2bf(x[i]);
}

__global__ void prep_state_kernel(unsigned short* __restrict__ h0,
                                  unsigned short* __restrict__ h1,
                                  unsigned* __restrict__ cnt) {
    int i = blockIdx.x * blockDim.x + threadIdx.x;
    if (i < 2 * Bb * Hh) { h0[i] = 0; h1[i] = 0; }
    if (i == 0) *cnt = 0;
}

// --------------------------- persistent LSTM kernel ------------------------

__global__ __launch_bounds__(THREADS) void lstm_persistent(
    const float* __restrict__ w_xh0, const float* __restrict__ b_xh0,
    const float* __restrict__ w_hh0, const float* __restrict__ b_hh0,
    const float* __restrict__ w_xh1, const float* __restrict__ b_xh1,
    const float* __restrict__ w_hh1, const float* __restrict__ b_hh1,
    const unsigned short* __restrict__ xbf,   // [B][T][I] bf16
    unsigned short* __restrict__ h0buf,       // [2][B][H] bf16 (double buffer)
    unsigned short* __restrict__ h1buf,       // [2][B][H] bf16
    unsigned* __restrict__ cnt,
    float* __restrict__ out)                  // outs | hT | cT
{
    __shared__ __attribute__((aligned(16))) unsigned short ldsw[LDS_E];

    const int tid = threadIdx.x;
    const int c0  = blockIdx.x * 16;          // this WG's H-column slice

    // ---- one-time: stage this WG's bf16 weight slice into LDS ----
    for (int e = tid; e < LDS_E; e += THREADS) {
        unsigned short v;
        if (e < WX0_E) {
            int q = e >> 12, rem = e & 4095, n = rem >> 8, k = rem & 255;
            v = f2bf(w_xh0[(size_t)(q * Hh + c0 + n) * Ii + k]);
        } else if (e < WX0_E + WH0_E) {
            int e2 = e - WX0_E;
            int q = e2 >> 13, rem = e2 & 8191, n = rem >> 9, k = rem & 511;
            v = f2bf(w_hh0[(size_t)(q * Hh + c0 + n) * Hh + k]);
        } else if (e < WX0_E + WH0_E + WX1_E) {
            int e2 = e - (WX0_E + WH0_E);
            int q = e2 >> 13, rem = e2 & 8191, n = rem >> 9, k = rem & 511;
            v = f2bf(w_xh1[(size_t)(q * Hh + c0 + n) * Hh + k]);
        } else {
            int e2 = e - (WX0_E + WH0_E + WX1_E);
            int q = e2 >> 13, rem = e2 & 8191, n = rem >> 9, k = rem & 511;
            v = f2bf(w_hh1[(size_t)(q * Hh + c0 + n) * Hh + k]);
        }
        ldsw[e] = v;
    }
    __syncthreads();

    const int wave  = tid >> 5;
    const int lane  = tid & 31;
    const int g     = lane >> 4;     // lane group for A/B fragment K-striping
    const int nl    = lane & 15;
    const int mbase = wave * 16;     // batch tile
    const int m_a   = mbase + nl;    // batch row this lane loads for A
    const int nh    = c0 + nl;       // H column this lane's D covers

    // combined biases per gate chunk (i,f,g,o)
    float bias0[4], bias1[4];
#pragma unroll
    for (int q = 0; q < 4; ++q) {
        bias0[q] = b_xh0[q * Hh + nh] + b_hh0[q * Hh + nh];
        bias1[q] = b_xh1[q * Hh + nh] + b_hh1[q * Hh + nh];
    }

    // per-lane LDS row base pointers (B-fragment rows = weight rows, n = nl)
    const unsigned short* wx0p[4];
    const unsigned short* wh0p[4];
    const unsigned short* wx1p[4];
    const unsigned short* wh1p[4];
#pragma unroll
    for (int q = 0; q < 4; ++q) {
        wx0p[q] = &ldsw[q * (16 * Ii) + nl * Ii];
        wh0p[q] = &ldsw[WX0_E + q * (16 * Hh) + nl * Hh];
        wx1p[q] = &ldsw[WX0_E + WH0_E + q * (16 * Hh) + nl * Hh];
        wh1p[q] = &ldsw[WX0_E + WH0_E + WX1_E + q * (16 * Hh) + nl * Hh];
    }

    v8f c_l0 = {};   // cell state, layer 0 (lives in registers all T steps)
    v8f c_l1 = {};   // cell state, layer 1
    unsigned target = 0;

    for (int t = 0; t < Tt; ++t) {
        const int cur = t & 1, prv = cur ^ 1;

        // ======================= layer 0 ===============================
        v8f acc[4];
#pragma unroll
        for (int q = 0; q < 4; ++q) {
            float bq = bias0[q];
#pragma unroll
            for (int r = 0; r < 8; ++r) acc[q][r] = bq;
        }
        // x(t) @ Wx0^T
        const unsigned short* arow = xbf + ((size_t)m_a * Tt + t) * Ii;
        for (int kc = 0; kc < Ii / 32; ++kc) {
            v16bf a = frag(arow + kc * 32, g);
#pragma unroll
            for (int q = 0; q < 4; ++q)
                acc[q] = wmma_bf16(a, frag(wx0p[q] + kc * 32, g), acc[q]);
        }
        // h0(t-1) @ Wh0^T
        const unsigned short* hrow = h0buf + (size_t)prv * (Bb * Hh) + (size_t)m_a * Hh;
        for (int kc = 0; kc < Hh / 32; ++kc) {
            v16bf a = frag(hrow + kc * 32, g);
#pragma unroll
            for (int q = 0; q < 4; ++q)
                acc[q] = wmma_bf16(a, frag(wh0p[q] + kc * 32, g), acc[q]);
        }
        // cell update (i,f,g,o) -> h0(t), c0(t)
        float hnew0[8];
#pragma unroll
        for (int r = 0; r < 8; ++r) {
            float iv = sigf(acc[0][r]);
            float fv = sigf(acc[1][r]);
            float gv = tanh_fast(acc[2][r]);
            float ov = sigf(acc[3][r]);
            float cc = c_l0[r] * fv + iv * gv;
            c_l0[r] = cc;
            hnew0[r] = ov * tanh_fast(cc);
        }
        unsigned short* h0w = h0buf + (size_t)cur * (Bb * Hh);
#pragma unroll
        for (int r = 0; r < 8; ++r) {
            int m = mbase + g * 8 + r;            // D-matrix row layout
            h0w[(size_t)m * Hh + nh] = f2bf(hnew0[r]);
        }
        target += NWG;
        grid_sync(cnt, target);

        // ======================= layer 1 ===============================
        v8f acc1[4];
#pragma unroll
        for (int q = 0; q < 4; ++q) {
            float bq = bias1[q];
#pragma unroll
            for (int r = 0; r < 8; ++r) acc1[q][r] = bq;
        }
        // h0(t) @ Wx1^T
        const unsigned short* a1 = h0buf + (size_t)cur * (Bb * Hh) + (size_t)m_a * Hh;
        for (int kc = 0; kc < Hh / 32; ++kc) {
            v16bf a = frag(a1 + kc * 32, g);
#pragma unroll
            for (int q = 0; q < 4; ++q)
                acc1[q] = wmma_bf16(a, frag(wx1p[q] + kc * 32, g), acc1[q]);
        }
        // h1(t-1) @ Wh1^T
        const unsigned short* a2 = h1buf + (size_t)prv * (Bb * Hh) + (size_t)m_a * Hh;
        for (int kc = 0; kc < Hh / 32; ++kc) {
            v16bf a = frag(a2 + kc * 32, g);
#pragma unroll
            for (int q = 0; q < 4; ++q)
                acc1[q] = wmma_bf16(a, frag(wh1p[q] + kc * 32, g), acc1[q]);
        }
        float hnew1[8];
#pragma unroll
        for (int r = 0; r < 8; ++r) {
            float iv = sigf(acc1[0][r]);
            float fv = sigf(acc1[1][r]);
            float gv = tanh_fast(acc1[2][r]);
            float ov = sigf(acc1[3][r]);
            float cc = c_l1[r] * fv + iv * gv;
            c_l1[r] = cc;
            hnew1[r] = ov * tanh_fast(cc);
        }
        unsigned short* h1w = h1buf + (size_t)cur * (Bb * Hh);
#pragma unroll
        for (int r = 0; r < 8; ++r) {
            int m = mbase + g * 8 + r;
            h1w[(size_t)m * Hh + nh] = f2bf(hnew1[r]);
            out[(size_t)m * Tt * Hh + (size_t)t * Hh + nh] = hnew1[r];
        }
        if (t == Tt - 1) {
            // final (hT, cT), layout (L=2, B, H) each, appended after outs
            const size_t ho = (size_t)Bb * Tt * Hh;
#pragma unroll
            for (int r = 0; r < 8; ++r) {
                int m = mbase + g * 8 + r;
                size_t idx = (size_t)m * Hh + nh;
                out[ho + 0 * (size_t)Bb * Hh + idx] = hnew0[r];
                out[ho + 1 * (size_t)Bb * Hh + idx] = hnew1[r];
                out[ho + 2 * (size_t)Bb * Hh + idx] = c_l0[r];
                out[ho + 3 * (size_t)Bb * Hh + idx] = c_l1[r];
            }
        }
        target += NWG;
        grid_sync(cnt, target);
    }
}

// ------------------------------- launcher ----------------------------------

extern "C" void kernel_launch(void* const* d_in, const int* in_sizes, int n_in,
                              void* d_out, int out_size, void* d_ws, size_t ws_size,
                              hipStream_t stream) {
    const float* x     = (const float*)d_in[0];
    const float* w_xh0 = (const float*)d_in[1];
    const float* b_xh0 = (const float*)d_in[2];
    const float* w_hh0 = (const float*)d_in[3];
    const float* b_hh0 = (const float*)d_in[4];
    const float* w_xh1 = (const float*)d_in[5];
    const float* b_xh1 = (const float*)d_in[6];
    const float* w_hh1 = (const float*)d_in[7];
    const float* b_hh1 = (const float*)d_in[8];
    float* out = (float*)d_out;

    // workspace layout
    char* ws = (char*)d_ws;
    unsigned* cnt       = (unsigned*)ws;                       // 256 B slot
    unsigned short* h0  = (unsigned short*)(ws + 256);         // [2][B][H] bf16
    unsigned short* h1  = h0 + 2 * Bb * Hh;                    // [2][B][H] bf16
    unsigned short* xbf = h1 + 2 * Bb * Hh;                    // [B][T][I] bf16

    const int nx = Bb * Tt * Ii;
    prep_x_kernel<<<(nx + 255) / 256, 256, 0, stream>>>(x, xbf, nx);
    prep_state_kernel<<<(2 * Bb * Hh + 255) / 256, 256, 0, stream>>>(h0, h1, cnt);

    lstm_persistent<<<NWG, THREADS, 0, stream>>>(
        w_xh0, b_xh0, w_hh0, b_hh0, w_xh1, b_xh1, w_hh1, b_hh1,
        xbf, h0, h1, cnt, out);
}
